// Attention_80642305950390
// MI455X (gfx1250) — matrix operand order repaired and checked by reading, not compiled, not run on hardware
//
#include <hip/hip_runtime.h>

// ---------------------------------------------------------------------------
// Fused single-head attention for MI455X (gfx1250), wave32 + WMMA f16.
//   B=4, S=2048, E=A=1024
// Phase 0: transpose+convert weights  W[E][A] f32 -> Wt[A][E] f16  (x3, tiny)
// Phase 1: Q/K projections row-major f16; V projection stored TRANSPOSED
//          (Vt[A][S] per batch) so every later WMMA operand is contiguous.
// Phase 2: scores = QK^T/sqrt(A), softmax, probs f16 (register-resident;
//          16 waves x 8 accumulators so nothing spills)
// Phase 3: out = P @ Vt (f16 x f16 -> f32)
// All intermediates (86 MB) live in d_ws -> L2-resident (192 MB L2).
// Every WMMA operand is loaded with 16B vector loads (no strided scalars).
// ---------------------------------------------------------------------------

typedef _Float16 half_t;
typedef __attribute__((ext_vector_type(16))) _Float16     v16h;
typedef __attribute__((ext_vector_type(8)))  _Float16     v8h;
typedef __attribute__((ext_vector_type(8)))  float        v8f;
typedef __attribute__((ext_vector_type(4)))  float        f4v;
typedef __attribute__((ext_vector_type(4)))  unsigned int uint4v;

static constexpr int Bb = 4;
static constexpr int Ss = 2048;
static constexpr int Ee = 1024;
static constexpr int Aa = 1024;

// -------------------- fragment loaders (wave32 WMMA layouts) ----------------
// A-matrix 16x32 f16: lane L: row = M0 + (L&15); lanes 0-15 hold K
// {0..7,16..23}, lanes 16-31 hold {8..15,24..31}.  f32 source, vectorized.
__device__ inline v16h load_a_f32(const float* __restrict__ X, int ld,
                                  int m0, int k0, int lane) {
  int row = m0 + (lane & 15);
  int kh  = (lane >> 4) * 8;
  const float* p = X + (size_t)row * ld + k0 + kh;
  f4v f0 = *(const f4v*)(p);
  f4v f1 = *(const f4v*)(p + 4);
  f4v f2 = *(const f4v*)(p + 16);
  f4v f3 = *(const f4v*)(p + 20);
  v16h a;
#pragma unroll
  for (int e = 0; e < 4; ++e) {
    a[e]      = (_Float16)f0[e];
    a[4 + e]  = (_Float16)f1[e];
    a[8 + e]  = (_Float16)f2[e];
    a[12 + e] = (_Float16)f3[e];
  }
  return a;
}

__device__ inline v16h load_a_f16(const half_t* __restrict__ X, int ld,
                                  int m0, int k0, int lane) {
  int row = m0 + (lane & 15);
  int kh  = (lane >> 4) * 8;
  const half_t* p = X + (size_t)row * ld + k0;
  union { v16h v; uint4v q[2]; } u;           // two 16B vector loads
  u.q[0] = *(const uint4v*)(p + kh);
  u.q[1] = *(const uint4v*)(p + 16 + kh);
  return u.v;
}

// B-operand 32(K)x16(N) built from a K-contiguous (transposed) f16 source:
// src is row-major [N_total][K_total]; B[k][n] = src[n][k].
// lane L: col n = n0+(L&15) -> row of src; K {0..7,16..23} / {8..15,24..31}.
__device__ inline v16h load_bT_f16(const half_t* __restrict__ T, int ld,
                                   int n0, int k0, int lane) {
  int row = n0 + (lane & 15);
  int kb  = (lane >> 4) * 8;
  const half_t* p = T + (size_t)row * ld + k0 + kb;
  union { v16h v; uint4v q[2]; } u;
  u.q[0] = *(const uint4v*)(p);
  u.q[1] = *(const uint4v*)(p + 16);
  return u.v;
}

// -------------------- Phase 0: W[E][A] f32 -> Wt[A][E] f16 ------------------
__global__ void __launch_bounds__(256)
transpose_w_f16(const float* __restrict__ W, half_t* __restrict__ Wt) {
  __shared__ float tile[32][33];
  int n0 = blockIdx.x * 32;     // A direction
  int k0 = blockIdx.y * 32;     // E direction
  int tx = threadIdx.x & 31;
  int ty = threadIdx.x >> 5;    // 8 rows of threads
#pragma unroll
  for (int r = ty; r < 32; r += 8)
    tile[r][tx] = W[(size_t)(k0 + r) * Aa + n0 + tx];
  __syncthreads();
#pragma unroll
  for (int r = ty; r < 32; r += 8)
    Wt[(size_t)(n0 + r) * Ee + k0 + tx] = (half_t)tile[tx][r];
}

// -------------------- Phase 1: X(f32) @ Wt(f16) + b ------------------------
// Block = 256 threads = 8 waves; wave tile 16x64 (4 wmma accumulators),
// block tile 64(M) x 128(N).  K = 1024, stepped by 32.
// TRANSPOSED_OUT=false: Out[m][n] row-major (Q, K) -- scalar b16 stores.
// TRANSPOSED_OUT=true : Out[b][n][s] (Vt) -- one packed b128 store per tile.
template <bool TRANSPOSED_OUT>
__global__ void __launch_bounds__(256)
qkv_proj_gemm(const float* __restrict__ X,
              const half_t* __restrict__ Wt,
              const float* __restrict__ bias,
              half_t* __restrict__ Out) {
  const int N = Aa, Kd = Ee;
  int tid  = threadIdx.x;
  int lane = tid & 31;
  int w    = tid >> 5;
  int wm   = w & 3;         // 4 waves along M
  int wn   = w >> 2;        // 2 waves along N
  int m0   = blockIdx.x * 64 + wm * 16;
  int nb   = blockIdx.y * 128 + wn * 64;

  v8f zero = {};
  v8f acc[4] = {zero, zero, zero, zero};

  for (int k0 = 0; k0 < Kd; k0 += 32) {
    if (k0 + 32 < Kd)  // pull next A strip toward the caches (global_prefetch_b8)
      __builtin_prefetch(X + (size_t)(m0 + (lane & 15)) * Kd + k0 + 32, 0, 3);
    v16h a = load_a_f32(X, Kd, m0, k0, lane);
#pragma unroll
    for (int t = 0; t < 4; ++t) {
      v16h bfrag = load_bT_f16(Wt, Kd, nb + t * 16, k0, lane);
      acc[t] = __builtin_amdgcn_wmma_f32_16x16x32_f16(
          false, a, false, bfrag, (short)0, acc[t], false, false);
    }
  }

  // C/D layout: lane: col = nb+t*16+(lane&15); vgpr v: row = m0 + mb + v.
  int ncol = lane & 15;
  int mb   = (lane >> 4) * 8;
  if (TRANSPOSED_OUT) {
    int b    = m0 >> 11;                 // batch from global row
    int mloc = (m0 & (Ss - 1)) + mb;     // row within batch (8-aligned)
    half_t* Ob = Out + (size_t)b * Ss * Aa;
#pragma unroll
    for (int t = 0; t < 4; ++t) {
      int col  = nb + t * 16 + ncol;
      float bv = bias[col];
      union { v8h h; uint4v q; } u;
#pragma unroll
      for (int v = 0; v < 8; ++v) u.h[v] = (half_t)(acc[t][v] + bv);
      *(uint4v*)(Ob + (size_t)col * Ss + mloc) = u.q;   // 8 contiguous halves
    }
  } else {
#pragma unroll
    for (int t = 0; t < 4; ++t) {
      int col  = nb + t * 16 + ncol;
      float bv = bias[col];
#pragma unroll
      for (int v = 0; v < 8; ++v)
        Out[(size_t)(m0 + mb + v) * N + col] = (half_t)(acc[t][v] + bv);
    }
  }
}

// -------------------- Phase 2: scores + softmax -> probs f16 ----------------
// One block per (batch, 16-query tile).  512 threads = 16 waves; each wave
// owns a 16x128 score chunk in registers (8 v8f accumulators = 64 VGPRs, no
// spills).  Row max/sum: shfl_xor within the half-wave (16 cols) + small LDS
// across the 16 waves.  launch_bounds(512,1) => maximum VGPR budget.
__global__ void __launch_bounds__(512, 1)
attn_scores_softmax(const half_t* __restrict__ Qw,
                    const half_t* __restrict__ Kw,
                    half_t* __restrict__ Pw) {
  int b    = blockIdx.y;
  int q0   = blockIdx.x * 16;
  int tid  = threadIdx.x;
  int lane = tid & 31;
  int w    = tid >> 5;                    // 0..15
  const half_t* Q  = Qw + (size_t)b * Ss * Aa;
  const half_t* Km = Kw + (size_t)b * Ss * Aa;
  half_t*       P  = Pw + (size_t)b * Ss * Ss;

  __shared__ float sred[16][16];

  v8f zero = {};
  v8f acc[8];
#pragma unroll
  for (int t = 0; t < 8; ++t) acc[t] = zero;

  int kw0 = w * 128;                      // this wave's 128-key span
  for (int a0 = 0; a0 < Aa; a0 += 32) {   // 32 k-steps over head dim
    v16h aq = load_a_f16(Q, Aa, q0, a0, lane);
#pragma unroll
    for (int t = 0; t < 8; ++t) {
      v16h bk = load_bT_f16(Km, Aa, kw0 + t * 16, a0, lane);
      acc[t] = __builtin_amdgcn_wmma_f32_16x16x32_f16(
          false, aq, false, bk, (short)0, acc[t], false, false);
    }
  }

  const float sc = 0.03125f;              // 1/sqrt(1024)
#pragma unroll
  for (int t = 0; t < 8; ++t) acc[t] = acc[t] * sc;

  int mb = (lane >> 4) * 8;               // row group within q-tile

  // ---- row max: over 8 tiles (per lane) then 16 cols (shfl) then 16 waves
  float gmax[8];
#pragma unroll
  for (int v = 0; v < 8; ++v) {
    float m = -3.0e38f;
#pragma unroll
    for (int t = 0; t < 8; ++t) m = fmaxf(m, acc[t][v]);
#pragma unroll
    for (int s = 1; s < 16; s <<= 1) m = fmaxf(m, __shfl_xor(m, s, 32));
    gmax[v] = m;
  }
  if ((lane & 15) == 0) {
#pragma unroll
    for (int v = 0; v < 8; ++v) sred[w][mb + v] = gmax[v];
  }
  __syncthreads();
#pragma unroll
  for (int v = 0; v < 8; ++v) {
    float m = sred[0][mb + v];
#pragma unroll
    for (int ww = 1; ww < 16; ++ww) m = fmaxf(m, sred[ww][mb + v]);
    gmax[v] = m;
  }
  __syncthreads();

  // ---- exp in place + row sum (same reduction tree)
  float rinv[8];
#pragma unroll
  for (int v = 0; v < 8; ++v) {
    float s = 0.f;
#pragma unroll
    for (int t = 0; t < 8; ++t) {
      float e = __expf(acc[t][v] - gmax[v]);
      acc[t][v] = e;
      s += e;
    }
#pragma unroll
    for (int sh = 1; sh < 16; sh <<= 1) s += __shfl_xor(s, sh, 32);
    rinv[v] = s;
  }
  if ((lane & 15) == 0) {
#pragma unroll
    for (int v = 0; v < 8; ++v) sred[w][mb + v] = rinv[v];
  }
  __syncthreads();
#pragma unroll
  for (int v = 0; v < 8; ++v) {
    float s = 0.f;
#pragma unroll
    for (int ww = 0; ww < 16; ++ww) s += sred[ww][mb + v];
    rinv[v] = 1.0f / s;
  }

  // ---- normalize + store probs as f16 (row-major: A-operand of PV GEMM)
  int ncol = lane & 15;
#pragma unroll
  for (int t = 0; t < 8; ++t) {
    int colb = kw0 + t * 16 + ncol;
#pragma unroll
    for (int v = 0; v < 8; ++v) {
      P[(size_t)(q0 + mb + v) * Ss + colb] = (half_t)(acc[t][v] * rinv[v]);
    }
  }
}

// -------------------- Phase 3: P(f16) @ Vt(f16) -> out f32 ------------------
// Vt is [A][S] per batch, so the B operand is K-contiguous (2 x b128 loads).
__global__ void __launch_bounds__(256)
pv_gemm(const half_t* __restrict__ Pw,
        const half_t* __restrict__ Vtw,
        float* __restrict__ Outw) {
  const int N = Aa, Kd = Ss;
  int b    = blockIdx.z;
  const half_t* Pm  = Pw  + (size_t)b * Ss * Ss;
  const half_t* Vt  = Vtw + (size_t)b * Ss * Aa;   // [A][S]
  float*        Out = Outw + (size_t)b * Ss * Aa;

  int tid  = threadIdx.x;
  int lane = tid & 31;
  int w    = tid >> 5;
  int wm   = w & 3;
  int wn   = w >> 2;
  int m0   = blockIdx.x * 64 + wm * 16;
  int nb   = blockIdx.y * 128 + wn * 64;

  v8f zero = {};
  v8f acc[4] = {zero, zero, zero, zero};

  for (int k0 = 0; k0 < Kd; k0 += 32) {
    if (k0 + 32 < Kd)
      __builtin_prefetch(Pm + (size_t)(m0 + (lane & 15)) * Kd + k0 + 32, 0, 3);
    v16h a = load_a_f16(Pm, Kd, m0, k0, lane);
#pragma unroll
    for (int t = 0; t < 4; ++t) {
      v16h bfrag = load_bT_f16(Vt, Ss, nb + t * 16, k0, lane);
      acc[t] = __builtin_amdgcn_wmma_f32_16x16x32_f16(
          false, a, false, bfrag, (short)0, acc[t], false, false);
    }
  }

  int ncol = lane & 15;
  int mb   = (lane >> 4) * 8;
#pragma unroll
  for (int t = 0; t < 4; ++t) {
    int col = nb + t * 16 + ncol;
#pragma unroll
    for (int v = 0; v < 8; ++v) {
      Out[(size_t)(m0 + mb + v) * N + col] = acc[t][v];
    }
  }
}

// ---------------------------------------------------------------------------
extern "C" void kernel_launch(void* const* d_in, const int* in_sizes, int n_in,
                              void* d_out, int out_size, void* d_ws, size_t ws_size,
                              hipStream_t stream) {
  (void)in_sizes; (void)n_in; (void)out_size; (void)ws_size;

  const float* query = (const float*)d_in[0];
  const float* key   = (const float*)d_in[1];
  const float* value = (const float*)d_in[2];
  const float* Wq    = (const float*)d_in[3];
  const float* bq    = (const float*)d_in[4];
  const float* Wk    = (const float*)d_in[5];
  const float* bk    = (const float*)d_in[6];
  const float* Wv    = (const float*)d_in[7];
  const float* bv    = (const float*)d_in[8];
  float* out = (float*)d_out;

  // Workspace (f16): Q | K | Vt | P | Wqt | Wkt | Wvt = 16+16+16+32+2+2+2 = 86 MB
  const size_t qkvElems = (size_t)Bb * Ss * Aa;          // 8M halves each
  const size_t pElems   = (size_t)Bb * Ss * Ss;          // 16M halves
  const size_t wElems   = (size_t)Ee * Aa;               // 1M halves each
  half_t* Qh  = (half_t*)d_ws;
  half_t* Kh  = Qh  + qkvElems;
  half_t* Vt  = Kh  + qkvElems;                          // [B][A][S]
  half_t* Ph  = Vt  + qkvElems;                          // [B][S][S]
  half_t* Wqt = Ph  + pElems;                            // [A][E]
  half_t* Wkt = Wqt + wElems;
  half_t* Wvt = Wkt + wElems;

  dim3 blk(256);

  // Phase 0: transpose+convert the three weight matrices (f32 -> f16)
  dim3 gtr(Aa / 32, Ee / 32);                            // (32, 32)
  transpose_w_f16<<<gtr, blk, 0, stream>>>(Wq, Wqt);
  transpose_w_f16<<<gtr, blk, 0, stream>>>(Wk, Wkt);
  transpose_w_f16<<<gtr, blk, 0, stream>>>(Wv, Wvt);

  // Phase 1: projections, M=B*S=8192, N=A=1024, K=E=1024
  dim3 gproj((Bb * Ss) / 64, Aa / 128);                  // (128, 8)
  qkv_proj_gemm<false><<<gproj, blk, 0, stream>>>(query, Wqt, bq, Qh);
  qkv_proj_gemm<false><<<gproj, blk, 0, stream>>>(key,   Wkt, bk, Kh);
  qkv_proj_gemm<true ><<<gproj, blk, 0, stream>>>(value, Wvt, bv, Vt);

  // Phase 2: scores + softmax -> probs f16 (512-thread blocks, 16 waves)
  dim3 gsc(Ss / 16, Bb);                                 // (128, 4)
  attn_scores_softmax<<<gsc, dim3(512), 0, stream>>>(Qh, Kh, Ph);

  // Phase 3: out = P @ Vt per batch, M=S=2048, N=A=1024, K=S=2048
  dim3 gpv(Ss / 64, Aa / 128, Bb);                       // (32, 8, 4)
  pv_gemm<<<gpv, blk, 0, stream>>>(Ph, Vt, out);
}
